// TransformerBlock_27608049779047
// MI455X (gfx1250) — matrix operand order repaired and checked by reading, not compiled, not run on hardware
//
#include <hip/hip_runtime.h>

// ---------------------------------------------------------------------------
// Types for CDNA5 WMMA (wave32): A/B = 16 bf16 per lane, C/D = 8 f32 per lane
// ---------------------------------------------------------------------------
typedef __bf16 bf16;
typedef __attribute__((ext_vector_type(16))) __bf16 v16bf;
typedef __attribute__((ext_vector_type(8)))  __bf16 v8bf;
typedef __attribute__((ext_vector_type(4)))  __bf16 v4bf;
typedef __attribute__((ext_vector_type(8)))  float  v8f;

#define EMBED  1024
#define T_SEQ  2048
#define NBATCH 2
#define NHEADS 16
#define HD     64
#define MROWS  4096   // NBATCH * T_SEQ
#define LDSPAD 40     // 32 + 8 bf16 row padding (80 B, 16 B aligned, bank-spread)

__device__ __forceinline__ v16bf cat8(v8bf lo, v8bf hi) {
  return __builtin_shufflevector(lo, hi, 0,1,2,3,4,5,6,7,8,9,10,11,12,13,14,15);
}

__device__ __forceinline__ v8f wmma_bf16(v16bf a, v16bf b, v8f c) {
  // D = A(16x32,bf16) * B(32x16,bf16) + C(16x16,f32)
  return __builtin_amdgcn_wmma_f32_16x16x32_bf16(false, a, false, b,
                                                 (short)0, c, false, false);
}

// ---------------------------------------------------------------------------
// Convert the six 1024x1024 fp32 weight matrices to bf16 (stays in L2)
// ---------------------------------------------------------------------------
__global__ __launch_bounds__(256) void cvt6_kernel(
    const float* __restrict__ s0, const float* __restrict__ s1,
    const float* __restrict__ s2, const float* __restrict__ s3,
    const float* __restrict__ s4, const float* __restrict__ s5,
    bf16* __restrict__ d0, bf16* __restrict__ d1, bf16* __restrict__ d2,
    bf16* __restrict__ d3, bf16* __restrict__ d4, bf16* __restrict__ d5,
    int n)
{
  int i = blockIdx.x * 256 + threadIdx.x;
  if (i < n) {
    d0[i] = (bf16)s0[i]; d1[i] = (bf16)s1[i]; d2[i] = (bf16)s2[i];
    d3[i] = (bf16)s3[i]; d4[i] = (bf16)s4[i]; d5[i] = (bf16)s5[i];
  }
}

// ---------------------------------------------------------------------------
// LayerNorm over rows of 1024, bf16 output. One block (256 thr) per row.
// ---------------------------------------------------------------------------
__global__ __launch_bounds__(256) void ln_kernel(
    const float* __restrict__ x, const float* __restrict__ g,
    const float* __restrict__ b, bf16* __restrict__ out)
{
  __shared__ float ssum[256];
  __shared__ float ssq[256];
  const int row = blockIdx.x;
  const int t   = threadIdx.x;
  const float4 v = ((const float4*)(x + (size_t)row * EMBED))[t];
  float s = v.x + v.y + v.z + v.w;
  float q = v.x*v.x + v.y*v.y + v.z*v.z + v.w*v.w;
  ssum[t] = s; ssq[t] = q;
  __syncthreads();
#pragma unroll
  for (int off = 128; off > 0; off >>= 1) {
    if (t < off) { ssum[t] += ssum[t+off]; ssq[t] += ssq[t+off]; }
    __syncthreads();
  }
  const float mu  = ssum[0] * (1.0f / EMBED);
  const float var = ssq[0] * (1.0f / EMBED) - mu * mu;
  const float rs  = rsqrtf(var + 1e-5f);
  const float4 gg = ((const float4*)g)[t];
  const float4 bb = ((const float4*)b)[t];
  v4bf o;
  o[0] = (bf16)((v.x - mu) * rs * gg.x + bb.x);
  o[1] = (bf16)((v.y - mu) * rs * gg.y + bb.y);
  o[2] = (bf16)((v.z - mu) * rs * gg.z + bb.z);
  o[3] = (bf16)((v.w - mu) * rs * gg.w + bb.w);
  ((v4bf*)(out + (size_t)row * EMBED))[t] = o;
}

// ---------------------------------------------------------------------------
// GEMM: Out = epilogue( A[4096x1024,bf16] * W[1024x1024,bf16]^T + bias )
// Block tile 64(M) x 256(N), 8 waves as 2x4, each wave 32x64 (8 WMMA accs).
// Double-buffered LDS staging (As 2x64x32, Bs 2x256x32, padded rows): each
// iteration prefetches the next k-slice into registers (VMEM overlaps WMMA),
// computes from the current buffer, then commits with ONE barrier.
// MODE 0: bf16 [M,N] (+ optional ReLU)
// MODE 1: bf16 head scatter  [B,H,T,64]  (q/k), value scaled
// MODE 2: bf16 head scatter transposed [B,H,64,T]  (v^T)
// MODE 3: fp32 [M,N], += residual
// ---------------------------------------------------------------------------
template <int MODE>
__global__ __launch_bounds__(256) void gemm16_kernel(
    const bf16* __restrict__ A, const bf16* __restrict__ W,
    const float* __restrict__ bias, const float* __restrict__ residual,
    float* __restrict__ outF, bf16* __restrict__ outB,
    float scale, int relu)
{
  __shared__ __align__(16) bf16 As[2][64][LDSPAD];     // 10 KB
  __shared__ __align__(16) bf16 Bs[2][256][LDSPAD];    // 40 KB
  const int K = EMBED, N = EMBED;
  const int tid  = threadIdx.x;
  const int NTN  = N / 256;                      // 4 column blocks
  const int m0   = (int)(blockIdx.x / NTN) * 64;
  const int n0   = (int)(blockIdx.x % NTN) * 256;
  const int w    = tid >> 5, lane = tid & 31;
  const int wm   = w & 1,  wn   = w >> 1;        // 2 x 4 wave grid
  const int lm   = lane & 15;
  const int hiHalf = lane >> 4;
  const int mOff = hiHalf << 3;

  // cooperative copy indices
  const int arow = tid >> 1, ahalf = tid & 1;    // threads 0..127: As segments
  const bf16* aSrc = A + (size_t)(m0 + arow) * K + ahalf * 16;
  const bf16* bSrc = W + (size_t)(n0 + tid) * K;           // row per thread

  v8f acc[2][4];
#pragma unroll
  for (int s = 0; s < 2; ++s)
#pragma unroll
    for (int j = 0; j < 4; ++j) acc[s][j] = 0.0f;

  // prologue: stage k-slice 0 into buffer 0
  {
    v8bf ra{}, rb0, rb1;
    if (tid < 128) ra = *(const v8bf*)aSrc;
    rb0 = *(const v8bf*)bSrc;
    rb1 = *(const v8bf*)(bSrc + 16);
    if (tid < 128) *(v8bf*)(&As[0][arow][ahalf * 16]) = ra;
    *(v8bf*)(&Bs[0][tid][0])  = rb0;
    *(v8bf*)(&Bs[0][tid][16]) = rb1;
  }
  __syncthreads();

  int cur = 0;
  for (int k0 = 0; k0 < K; k0 += 32) {
    const bool hasNext = (k0 + 32) < K;
    // prefetch next slice into registers (loads issue before the WMMA block)
    v8bf ra{}, rb0{}, rb1{};
    if (hasNext) {
      if (tid < 128) ra = *(const v8bf*)(aSrc + k0 + 32);
      rb0 = *(const v8bf*)(bSrc + k0 + 32);
      rb1 = *(const v8bf*)(bSrc + k0 + 48);
    }

    // compute from current buffer
    v16bf a[2];
#pragma unroll
    for (int s = 0; s < 2; ++s) {
      const bf16* ap = &As[cur][wm * 32 + s * 16 + lm][hiHalf << 3];
      a[s] = cat8(*(const v8bf*)ap, *(const v8bf*)(ap + 16));
    }
#pragma unroll
    for (int j = 0; j < 4; ++j) {
      const bf16* bp = &Bs[cur][wn * 64 + j * 16 + lm][hiHalf << 3];
      const v16bf b = cat8(*(const v8bf*)bp, *(const v8bf*)(bp + 16));
#pragma unroll
      for (int s = 0; s < 2; ++s)
        acc[s][j] = wmma_bf16(a[s], b, acc[s][j]);
    }

    if (hasNext) {
      const int nxt = cur ^ 1;
      if (tid < 128) *(v8bf*)(&As[nxt][arow][ahalf * 16]) = ra;
      *(v8bf*)(&Bs[nxt][tid][0])  = rb0;
      *(v8bf*)(&Bs[nxt][tid][16]) = rb1;
      __syncthreads();          // one barrier per k-step
      cur = nxt;
    }
  }

#pragma unroll
  for (int s = 0; s < 2; ++s) {
#pragma unroll
    for (int j = 0; j < 4; ++j) {
      const int col = n0 + wn * 64 + j * 16 + lm;
      const float bv = bias[col];
#pragma unroll
      for (int r = 0; r < 8; ++r) {
        const int row = m0 + wm * 32 + s * 16 + r + mOff;
        float val = (acc[s][j][r] + bv) * scale;
        if (relu) val = fmaxf(val, 0.0f);
        if (MODE == 0) {
          outB[(size_t)row * N + col] = (bf16)val;
        } else if (MODE == 1) {
          const int b_ = row >> 11, t_ = row & (T_SEQ - 1);
          const int h_ = col >> 6,  d_ = col & (HD - 1);
          outB[(((size_t)(b_ * NHEADS + h_) * T_SEQ) + t_) * HD + d_] = (bf16)val;
        } else if (MODE == 2) {
          const int b_ = row >> 11, t_ = row & (T_SEQ - 1);
          const int h_ = col >> 6,  d_ = col & (HD - 1);
          outB[((size_t)(b_ * NHEADS + h_) * HD + d_) * T_SEQ + t_] = (bf16)val;
        } else {
          outF[(size_t)row * N + col] =
              residual[(size_t)row * N + col] + val;
        }
      }
    }
  }
}

// ---------------------------------------------------------------------------
// Causal flash attention. q:[B,H,T,64] (pre-scaled), k:[B,H,T,64],
// vt:[B,H,64,T]. One wave owns a 16-query tile; loops over 32-key tiles with
// online softmax. P converts C-layout -> A-layout via a wave-private LDS tile.
// ctx out: [B,T,1024] bf16.
// ---------------------------------------------------------------------------
__global__ __launch_bounds__(256) void attn_kernel(
    const bf16* __restrict__ Q, const bf16* __restrict__ Kmat,
    const bf16* __restrict__ Vt, bf16* __restrict__ ctx)
{
  __shared__ __align__(16) bf16 plds[8][16][LDSPAD];   // per-wave 16x32 P tile
  const int w    = threadIdx.x >> 5;
  const int lane = threadIdx.x & 31;
  const int qb   = blockIdx.x & 15;      // 16 blocks of 128 queries
  const int bh   = blockIdx.x >> 4;      // b*16 + h, 0..31
  const int q0   = qb * 128 + w * 16;

  const bf16* qh = Q    + (size_t)bh * T_SEQ * HD;
  const bf16* kh = Kmat + (size_t)bh * T_SEQ * HD;
  const bf16* vh = Vt   + (size_t)bh * HD * T_SEQ;

  const int lm = lane & 15;
  const int hiHalf = lane >> 4;
  const int mOff = hiHalf << 3;

  // Q fragments (d = 0..31 and 32..63)
  const bf16* qrow = qh + (size_t)(q0 + lm) * HD + (hiHalf << 4);
  const v16bf a0 = cat8(*(const v8bf*)(qrow),      *(const v8bf*)(qrow + 8));
  const v16bf a1 = cat8(*(const v8bf*)(qrow + 32), *(const v8bf*)(qrow + 40));

  v8f o[4];
#pragma unroll
  for (int nd = 0; nd < 4; ++nd) o[nd] = 0.0f;
  float rmax[8], rsum[8];
#pragma unroll
  for (int r = 0; r < 8; ++r) { rmax[r] = -1e30f; rsum[r] = 0.0f; }

  const int jmax = (q0 + 15) >> 5;
  for (int j = 0; j <= jmax; ++j) {
    const int kb = j * 32;

    // S = Q * K^T  (two 16x16 score fragments: keys kb..+15 and kb+16..+31)
    v8f s0 = 0.0f, s1 = 0.0f;
    {
      const bf16* kr = kh + (size_t)(kb + lm) * HD + (hiHalf << 4);
      s0 = wmma_bf16(a0, cat8(*(const v8bf*)kr, *(const v8bf*)(kr + 8)), s0);
      s0 = wmma_bf16(a1, cat8(*(const v8bf*)(kr + 32), *(const v8bf*)(kr + 40)), s0);
      const bf16* kr2 = kh + (size_t)(kb + 16 + lm) * HD + (hiHalf << 4);
      s1 = wmma_bf16(a0, cat8(*(const v8bf*)kr2, *(const v8bf*)(kr2 + 8)), s1);
      s1 = wmma_bf16(a1, cat8(*(const v8bf*)(kr2 + 32), *(const v8bf*)(kr2 + 40)), s1);
    }

    // Causal mask + online softmax (fp32). Row stats live replicated in the
    // 16-lane half that owns that row of the C fragment.
#pragma unroll
    for (int r = 0; r < 8; ++r) {
      const int qrowi = q0 + r + mOff;
      float v0 = (kb + lm      <= qrowi) ? s0[r] : -1e30f;
      float v1 = (kb + 16 + lm <= qrowi) ? s1[r] : -1e30f;
      float mt = fmaxf(v0, v1);
#pragma unroll
      for (int off = 8; off >= 1; off >>= 1)
        mt = fmaxf(mt, __shfl_xor(mt, off, 16));
      const float mnew = fmaxf(rmax[r], mt);
      const float corr = __expf(rmax[r] - mnew);
      const float p0 = __expf(v0 - mnew);
      const float p1 = __expf(v1 - mnew);
      float ps = p0 + p1;
#pragma unroll
      for (int off = 8; off >= 1; off >>= 1)
        ps += __shfl_xor(ps, off, 16);
      rsum[r] = rsum[r] * corr + ps;
      rmax[r] = mnew;
#pragma unroll
      for (int nd = 0; nd < 4; ++nd) o[nd][r] *= corr;
      plds[w][r + mOff][lm]      = (bf16)p0;
      plds[w][r + mOff][lm + 16] = (bf16)p1;
    }

    // Reload P as an A fragment (16x32) from the wave-private LDS tile.
    const bf16* prow = &plds[w][lm][0];
    const v16bf p = cat8(*(const v8bf*)(prow + (hiHalf << 4)),
                         *(const v8bf*)(prow + 8 + (hiHalf << 4)));

    // O += P * V   (B operand rows are contiguous thanks to [B,H,64,T] v^T)
#pragma unroll
    for (int nd = 0; nd < 4; ++nd) {
      const bf16* vrow = vh + (size_t)(nd * 16 + lm) * T_SEQ + kb + (hiHalf << 4);
      const v16bf bv = cat8(*(const v8bf*)vrow, *(const v8bf*)(vrow + 8));
      o[nd] = wmma_bf16(p, bv, o[nd]);
    }
  }

  // Normalize and scatter to ctx [B,T,1024]
  const int b_ = bh >> 4, h_ = bh & 15;
#pragma unroll
  for (int r = 0; r < 8; ++r) {
    const float inv = 1.0f / rsum[r];
    const int t_ = q0 + r + mOff;
    bf16* crow = ctx + ((size_t)(b_ * T_SEQ + t_)) * EMBED + h_ * HD;
#pragma unroll
    for (int nd = 0; nd < 4; ++nd)
      crow[nd * 16 + lm] = (bf16)(o[nd][r] * inv);
  }
}

// ---------------------------------------------------------------------------
// Host launcher
// ---------------------------------------------------------------------------
extern "C" void kernel_launch(void* const* d_in, const int* in_sizes, int n_in,
                              void* d_out, int out_size, void* d_ws, size_t ws_size,
                              hipStream_t stream) {
  const float* x  = (const float*)d_in[0];
  const float* Wq = (const float*)d_in[1];  const float* bq = (const float*)d_in[2];
  const float* Wk = (const float*)d_in[3];  const float* bk = (const float*)d_in[4];
  const float* Wv = (const float*)d_in[5];  const float* bv = (const float*)d_in[6];
  const float* Wo = (const float*)d_in[7];  const float* bo = (const float*)d_in[8];
  const float* W1 = (const float*)d_in[9];  const float* b1 = (const float*)d_in[10];
  const float* W2 = (const float*)d_in[11]; const float* b2 = (const float*)d_in[12];
  const float* g1 = (const float*)d_in[13]; const float* be1 = (const float*)d_in[14];
  const float* g2 = (const float*)d_in[15]; const float* be2 = (const float*)d_in[16];

  char* ws = (char*)d_ws;
  const size_t WB  = (size_t)EMBED * EMBED * sizeof(bf16);      // 2 MB
  const size_t ACT = (size_t)MROWS * EMBED * sizeof(bf16);      // 8 MB
  size_t off = 0;
  bf16* Wq_b = (bf16*)(ws + off); off += WB;
  bf16* Wk_b = (bf16*)(ws + off); off += WB;
  bf16* Wv_b = (bf16*)(ws + off); off += WB;
  bf16* Wo_b = (bf16*)(ws + off); off += WB;
  bf16* W1_b = (bf16*)(ws + off); off += WB;
  bf16* W2_b = (bf16*)(ws + off); off += WB;
  bf16* bufA = (bf16*)(ws + off); off += ACT;   // xn1, later ctx
  bf16* bufB = (bf16*)(ws + off); off += ACT;   // q,   later xn2
  bf16* bufC = (bf16*)(ws + off); off += ACT;   // k,   later h1
  bf16* bufD = (bf16*)(ws + off); off += ACT;   // v^T
  float* x1  = (float*)(ws + off); off += (size_t)MROWS * EMBED * sizeof(float);

  const dim3 blk(256);
  const dim3 gridCvt(4096);     // 1M elements / 256
  const dim3 gridLN(MROWS);     // one row per block
  const dim3 gridGemm(256);     // (4096/64) * (1024/256) block tiles
  const dim3 gridAttn(512);     // B*H * (T/128)

  // 1) weights -> bf16
  cvt6_kernel<<<gridCvt, blk, 0, stream>>>(Wq, Wk, Wv, Wo, W1, W2,
                                           Wq_b, Wk_b, Wv_b, Wo_b, W1_b, W2_b,
                                           EMBED * EMBED);
  // 2) xn1 = LN1(x)
  ln_kernel<<<gridLN, blk, 0, stream>>>(x, g1, be1, bufA);
  // 3) q = (xn1 Wq^T + bq) * d^-0.5, head layout
  gemm16_kernel<1><<<gridGemm, blk, 0, stream>>>(bufA, Wq_b, bq, nullptr,
                                                 nullptr, bufB, 0.125f, 0);
  // 4) k, head layout
  gemm16_kernel<1><<<gridGemm, blk, 0, stream>>>(bufA, Wk_b, bk, nullptr,
                                                 nullptr, bufC, 1.0f, 0);
  // 5) v^T, transposed head layout
  gemm16_kernel<2><<<gridGemm, blk, 0, stream>>>(bufA, Wv_b, bv, nullptr,
                                                 nullptr, bufD, 1.0f, 0);
  // 6) causal attention -> ctx (reuses bufA)
  attn_kernel<<<gridAttn, blk, 0, stream>>>(bufB, bufC, bufD, bufA);
  // 7) x1 = x + ctx Wo^T + bo
  gemm16_kernel<3><<<gridGemm, blk, 0, stream>>>(bufA, Wo_b, bo, x,
                                                 x1, nullptr, 1.0f, 0);
  // 8) xn2 = LN2(x1)
  ln_kernel<<<gridLN, blk, 0, stream>>>(x1, g2, be2, bufB);
  // 9) h1 = relu(xn2 W1^T + b1)
  gemm16_kernel<0><<<gridGemm, blk, 0, stream>>>(bufB, W1_b, b1, nullptr,
                                                 nullptr, bufC, 1.0f, 1);
  // 10) out = x1 + h1 W2^T + b2
  gemm16_kernel<3><<<gridGemm, blk, 0, stream>>>(bufC, W2_b, b2, x1,
                                                 (float*)d_out, nullptr, 1.0f, 0);
}
